// GraphSAGE_16716012716699
// MI455X (gfx1250) — compile-verified, matrix operand order checked by reading, not run on hardware
//
#include <hip/hip_runtime.h>
#include <hip/hip_bf16.h>

// ---- problem constants (from reference) ----
#define N_NODES 50000
#define N_EDGES 800000
#define IN_DIM  96
#define HID     32
#define OUTD    64

typedef __attribute__((ext_vector_type(16))) __bf16 v16bf;
typedef __attribute__((ext_vector_type(8)))  float  v8f;

union FragU { uint4 u[2]; v16bf v; };   // 32-byte bit-pun helper

// K-index held by element e of an A/B bf16 fragment, for lane-half hl (0/1).
// ISA 16-bit 16x32 layout: VGPR0-3 hold K=0..15 (split by lane half),
// VGPR4-7 hold K=16..31.  => K = e + 8*hl + (e>=8 ? 8 : 0)
__device__ __forceinline__ int kmap(int e, int hl) {
    return e + 8 * hl + (e >= 8 ? 8 : 0);
}

// ---------------- degree count ----------------
__global__ __launch_bounds__(256) void degree_kernel(const int* __restrict__ edst,
                                                     float* __restrict__ cnt) {
    int e = blockIdx.x * 256 + threadIdx.x;
    if (e < N_EDGES) atomicAdd(&cnt[edst[e]], 1.0f);
}

// ---------------- layer-1 GEMM: [P1|R1] = bf16(x) @ [W1l|W1r] ----------------
// x: [N,96] f32.  W1l/W1r: [96,32] f32.  P1,R1: [N,32] f32.
// One wave computes a 16-row x 64-col tile: 4 accumulators, K=96 (3 steps).
// B fragments are pre-packed into LDS in per-lane register layout so each
// lane reads its whole v16bf fragment as two ds_load_b128.
__global__ __launch_bounds__(256) void gemm1_kernel(const float* __restrict__ x,
                                                    const float* __restrict__ W1l,
                                                    const float* __restrict__ W1r,
                                                    float* __restrict__ P1,
                                                    float* __restrict__ R1) {
    // 12 fragments (3 ksteps x 4 ntiles) x 32 lanes x 16 bf16 = 12 KB
    __shared__ __align__(32) __bf16 frag[12 * 32 * 16];
    int t = threadIdx.x;
    for (int i = t; i < 12 * 32 * 16; i += 256) {
        int f    = i >> 9;           // fragment id
        int lane = (i >> 4) & 31;
        int e    = i & 15;
        int ks   = f >> 2, nt = f & 3;
        int hl   = lane >> 4, lm = lane & 15;
        int K    = ks * 32 + kmap(e, hl);
        int col  = nt * 16 + lm;     // 0-31 -> W1l, 32-63 -> W1r
        float v  = (col < 32) ? W1l[K * 32 + col] : W1r[K * 32 + (col - 32)];
        frag[i]  = (__bf16)v;
    }
    __syncthreads();

    int wave = t >> 5;
    int lane = t & 31;
    int hl   = lane >> 4;
    int lm   = lane & 15;
    int row0 = (blockIdx.x * 8 + wave) * 16;       // 3125 tiles (50000/16)
    if (row0 >= N_NODES) return;                   // wave-uniform: EXEC all-1s

    const v16bf* bfrag = (const v16bf*)frag;
    v8f acc[4] = {v8f{}, v8f{}, v8f{}, v8f{}};
    const float* xr = x + (size_t)(row0 + lm) * IN_DIM;

    #pragma unroll
    for (int ks = 0; ks < 3; ++ks) {
        int k0 = ks * 32;
        // A fragment: two contiguous 8-float K-runs, 16B-aligned b128 loads
        const float* base = xr + k0 + 8 * hl;
        float4 q0 = *(const float4*)(base);
        float4 q1 = *(const float4*)(base + 4);
        float4 q2 = *(const float4*)(base + 16);
        float4 q3 = *(const float4*)(base + 20);
        v16bf a;
        a[0] = (__bf16)q0.x;  a[1] = (__bf16)q0.y;  a[2]  = (__bf16)q0.z;  a[3]  = (__bf16)q0.w;
        a[4] = (__bf16)q1.x;  a[5] = (__bf16)q1.y;  a[6]  = (__bf16)q1.z;  a[7]  = (__bf16)q1.w;
        a[8] = (__bf16)q2.x;  a[9] = (__bf16)q2.y;  a[10] = (__bf16)q2.z;  a[11] = (__bf16)q2.w;
        a[12] = (__bf16)q3.x; a[13] = (__bf16)q3.y; a[14] = (__bf16)q3.z;  a[15] = (__bf16)q3.w;
        #pragma unroll
        for (int nt = 0; nt < 4; ++nt) {
            v16bf b = bfrag[(ks * 4 + nt) * 32 + lane];   // 32B contiguous LDS read
            acc[nt] = __builtin_amdgcn_wmma_f32_16x16x32_bf16(
                false, a, false, b, (short)0, acc[nt], false, false);
        }
    }
    // C layout: VGPR r -> M = r + 8*hl, N = lm
    #pragma unroll
    for (int nt = 0; nt < 4; ++nt) {
        float* dst = (nt < 2) ? P1 : R1;
        int c0 = (nt & 1) * 16;
        #pragma unroll
        for (int r = 0; r < 8; ++r) {
            int row = row0 + r + 8 * hl;
            dst[(size_t)row * HID + c0 + lm] = acc[nt][r];
        }
    }
}

// ---------------- edge scatter-add, 32 dims ----------------
__global__ __launch_bounds__(256) void scatter32_kernel(const int* __restrict__ esrc,
                                                        const int* __restrict__ edst,
                                                        const float* __restrict__ P,
                                                        float* __restrict__ agg) {
    unsigned tid = blockIdx.x * 256 + threadIdx.x;
    unsigned e = tid >> 5;       // uniform within wave32 -> scalar edge reads
    unsigned d = tid & 31;
    if (e < N_EDGES)
        atomicAdd(&agg[(size_t)edst[e] * HID + d], P[(size_t)esrc[e] * HID + d]);
}

// ---------------- finalize layer 1: mean + root + bias, ReLU -> bf16 h ----------------
__global__ __launch_bounds__(256) void finalize1_kernel(const float* __restrict__ agg,
                                                        const float* __restrict__ R,
                                                        const float* __restrict__ cnt,
                                                        const float* __restrict__ b1,
                                                        __bf16* __restrict__ h) {
    unsigned tid = blockIdx.x * 256 + threadIdx.x;     // N*8 threads, 4 dims each
    if (tid >= (unsigned)N_NODES * 8) return;
    unsigned i  = tid >> 3;
    unsigned d4 = (tid & 7) * 4;
    float inv = 1.0f / fmaxf(cnt[i], 1.0f);
    float4 av = *(const float4*)(agg + (size_t)i * HID + d4);
    float4 rv = *(const float4*)(R   + (size_t)i * HID + d4);
    float4 bv = *(const float4*)(b1 + d4);
    union { __bf16 q[4]; uint2 u; } o;
    o.q[0] = (__bf16)fmaxf(av.x * inv + rv.x + bv.x, 0.0f);
    o.q[1] = (__bf16)fmaxf(av.y * inv + rv.y + bv.y, 0.0f);
    o.q[2] = (__bf16)fmaxf(av.z * inv + rv.z + bv.z, 0.0f);
    o.q[3] = (__bf16)fmaxf(av.w * inv + rv.w + bv.w, 0.0f);
    *(uint2*)(h + (size_t)i * HID + d4) = o.u;
}

// ---------------- layer-2 GEMM: [P2|R2] = h @ [W2l|W2r] ----------------
// h: [N,32] bf16.  W2l/W2r: [32,64] f32.  P2,R2: [N,64] f32.
// One wave computes 16 rows x 128 cols: 8 accumulators, single K=32 step.
__global__ __launch_bounds__(256) void gemm2_kernel(const __bf16* __restrict__ h,
                                                    const float* __restrict__ W2l,
                                                    const float* __restrict__ W2r,
                                                    float* __restrict__ P2,
                                                    float* __restrict__ R2) {
    // 8 fragments x 32 lanes x 16 bf16 = 8 KB
    __shared__ __align__(32) __bf16 frag[8 * 32 * 16];
    int t = threadIdx.x;
    for (int i = t; i < 8 * 32 * 16; i += 256) {
        int nt   = i >> 9;
        int lane = (i >> 4) & 31;
        int e    = i & 15;
        int hl   = lane >> 4, lm = lane & 15;
        int K    = kmap(e, hl);
        int col  = nt * 16 + lm;     // 0-63 -> W2l, 64-127 -> W2r
        float v  = (col < 64) ? W2l[K * 64 + col] : W2r[K * 64 + (col - 64)];
        frag[i]  = (__bf16)v;
    }
    __syncthreads();

    int wave = t >> 5;
    int lane = t & 31;
    int hl   = lane >> 4;
    int lm   = lane & 15;
    int row0 = (blockIdx.x * 8 + wave) * 16;
    if (row0 >= N_NODES) return;

    // A fragment: packed bf16 already in K order -> two 16B loads, bit-pun
    const __bf16* hr = h + (size_t)(row0 + lm) * HID;
    FragU au;
    au.u[0] = *(const uint4*)(hr + 8 * hl);        // elements 0-7  (K = 8hl..8hl+7)
    au.u[1] = *(const uint4*)(hr + 16 + 8 * hl);   // elements 8-15 (K = 16+8hl..)
    v16bf a = au.v;

    const v16bf* bfrag = (const v16bf*)frag;
    v8f acc[8] = {v8f{}, v8f{}, v8f{}, v8f{}, v8f{}, v8f{}, v8f{}, v8f{}};
    #pragma unroll
    for (int nt = 0; nt < 8; ++nt) {
        v16bf b = bfrag[nt * 32 + lane];
        acc[nt] = __builtin_amdgcn_wmma_f32_16x16x32_bf16(
            false, a, false, b, (short)0, acc[nt], false, false);
    }
    #pragma unroll
    for (int nt = 0; nt < 8; ++nt) {
        float* dst = (nt < 4) ? P2 : R2;
        int c0 = (nt & 3) * 16;
        #pragma unroll
        for (int r = 0; r < 8; ++r) {
            int row = row0 + r + 8 * hl;
            dst[(size_t)row * OUTD + c0 + lm] = acc[nt][r];
        }
    }
}

// ---------------- edge scatter-add, 64 dims ----------------
__global__ __launch_bounds__(256) void scatter64_kernel(const int* __restrict__ esrc,
                                                        const int* __restrict__ edst,
                                                        const float* __restrict__ P,
                                                        float* __restrict__ agg) {
    unsigned tid = blockIdx.x * 256 + threadIdx.x;
    unsigned e = tid >> 6;
    unsigned d = tid & 63;
    if (e < N_EDGES)
        atomicAdd(&agg[(size_t)edst[e] * OUTD + d], P[(size_t)esrc[e] * OUTD + d]);
}

// ---------------- finalize layer 2 -> fp32 output ----------------
__global__ __launch_bounds__(256) void finalize2_kernel(const float* __restrict__ agg,
                                                        const float* __restrict__ R,
                                                        const float* __restrict__ cnt,
                                                        const float* __restrict__ b2,
                                                        float* __restrict__ out) {
    unsigned tid = blockIdx.x * 256 + threadIdx.x;     // N*16 threads, 4 dims each
    if (tid >= (unsigned)N_NODES * 16) return;
    unsigned i  = tid >> 4;
    unsigned d4 = (tid & 15) * 4;
    float inv = 1.0f / fmaxf(cnt[i], 1.0f);
    float4 av = *(const float4*)(agg + (size_t)i * OUTD + d4);
    float4 rv = *(const float4*)(R   + (size_t)i * OUTD + d4);
    float4 bv = *(const float4*)(b2 + d4);
    float4 o;
    o.x = av.x * inv + rv.x + bv.x;
    o.y = av.y * inv + rv.y + bv.y;
    o.z = av.z * inv + rv.z + bv.z;
    o.w = av.w * inv + rv.w + bv.w;
    *(float4*)(out + (size_t)i * OUTD + d4) = o;
}

extern "C" void kernel_launch(void* const* d_in, const int* in_sizes, int n_in,
                              void* d_out, int out_size, void* d_ws, size_t ws_size,
                              hipStream_t stream) {
    const float* x    = (const float*)d_in[0];
    const int*   edge = (const int*)d_in[1];      // [2,E] int32 (JAX x64 off)
    const int*   esrc = edge;
    const int*   edst = edge + N_EDGES;
    const float* W1l  = (const float*)d_in[2];
    const float* W1r  = (const float*)d_in[3];
    const float* b1   = (const float*)d_in[4];
    const float* W2l  = (const float*)d_in[5];
    const float* W2r  = (const float*)d_in[6];
    const float* b2   = (const float*)d_in[7];

    // workspace layout (byte offsets, all multiples of 4*N -> 16B aligned)
    const size_t N = N_NODES;
    char* ws = (char*)d_ws;
    float*  cnt  = (float*)(ws);                  // N f32
    float*  P1   = (float*)(ws + 4   * N);        // 32N f32
    float*  R1   = (float*)(ws + 132 * N);        // 32N f32
    float*  agg1 = (float*)(ws + 260 * N);        // 32N f32
    __bf16* h    = (__bf16*)(ws + 388 * N);       // 32N bf16 (64N bytes)
    float*  agg2 = (float*)(ws + 452 * N);        // 64N f32
    float*  R2   = (float*)(ws + 708 * N);        // 64N f32  (total 964N B ~ 48MB)
    float*  P2   = P1;                            // reuse dead P1|R1 region (64N f32)

    hipMemsetAsync(cnt,  0, 4   * N, stream);
    hipMemsetAsync(agg1, 0, 128 * N, stream);
    hipMemsetAsync(agg2, 0, 256 * N, stream);

    const int TILES = N_NODES / 16;               // 3125 exactly
    const int GBLKS = (TILES + 7) / 8;            // 8 wave-tiles per block

    degree_kernel   <<<(N_EDGES + 255) / 256, 256, 0, stream>>>(edst, cnt);
    gemm1_kernel    <<<GBLKS, 256, 0, stream>>>(x, W1l, W1r, P1, R1);
    scatter32_kernel<<<((size_t)N_EDGES * 32 + 255) / 256, 256, 0, stream>>>(esrc, edst, P1, agg1);
    finalize1_kernel<<<((size_t)N_NODES * 8 + 255) / 256, 256, 0, stream>>>(agg1, R1, cnt, b1, h);
    gemm2_kernel    <<<GBLKS, 256, 0, stream>>>(h, W2l, W2r, P2, R2);
    scatter64_kernel<<<((size_t)N_EDGES * 64 + 255) / 256, 256, 0, stream>>>(esrc, edst, P2, agg2);
    finalize2_kernel<<<((size_t)N_NODES * 16 + 255) / 256, 256, 0, stream>>>(agg2, R2, cnt, b2, (float*)d_out);
}
